// FABlock3D_27015344292360
// MI455X (gfx1250) — matrix operand order repaired and checked by reading, not compiled, not run on hardware
//
#include <hip/hip_runtime.h>
#include <math.h>

typedef __attribute__((ext_vector_type(16))) __bf16 v16bf;
typedef __attribute__((ext_vector_type(8)))  float  v8f;

#define DIMC 128
#define NPTS (48*48*48)      /* 110592 spatial points */
#define HC   512             /* HEADS*DH */

__device__ __forceinline__ float  bf2f(__bf16 x) { return (float)x; }
__device__ __forceinline__ __bf16 f2bf(float x)  { return (__bf16)x; }
__device__ __forceinline__ float gelu_exact(float x) {
  return 0.5f * x * (1.0f + erff(x * 0.70710678118654752f));
}

// ---------------------------------------------------------------- utilities
__global__ void k_zero(float* p, int n) {
  int i = blockIdx.x * blockDim.x + threadIdx.x;
  if (i < n) p[i] = 0.f;
}

// in: (K,N) f32 row-major -> out: (N,K) bf16 row-major (transposed weights)
__global__ void k_convT(const float* __restrict__ in, __bf16* __restrict__ out,
                        int K, int N) {
  int i = blockIdx.x * blockDim.x + threadIdx.x;
  if (i >= K * N) return;
  int k = i / N, n = i - k * N;
  out[(size_t)n * K + k] = f2bf(in[i]);
}

// ------------------------------------------------- LayerNorm + axis pooling
// one block (128 threads) per spatial point; writes ln(u) in bf16 and
// accumulates the three pooled means (mean commutes with the later linears)
__global__ void k_ln_pool(const float* __restrict__ u, const float* __restrict__ lnw,
                          const float* __restrict__ lnb, __bf16* __restrict__ lnU,
                          float* poolx, float* pooly, float* poolz) {
  __shared__ float red[128];
  int row = blockIdx.x, c = threadIdx.x;
  float x = u[(size_t)row * DIMC + c];
  red[c] = x; __syncthreads();
  for (int s = 64; s > 0; s >>= 1) { if (c < s) red[c] += red[c + s]; __syncthreads(); }
  float mean = red[0] * (1.0f / DIMC); __syncthreads();
  float d = x - mean;
  red[c] = d * d; __syncthreads();
  for (int s = 64; s > 0; s >>= 1) { if (c < s) red[c] += red[c + s]; __syncthreads(); }
  float var = red[0] * (1.0f / DIMC);
  float y = d * rsqrtf(var + 1e-5f) * lnw[c] + lnb[c];
  lnU[(size_t)row * DIMC + c] = f2bf(y);
  int zi = row % 48, yi = (row / 48) % 48, xi = row / 2304;
  float yw = y * (1.0f / 2304.0f);
  atomicAdd(&poolx[xi * DIMC + c], yw);
  atomicAdd(&pooly[yi * DIMC + c], yw);
  atomicAdd(&poolz[zi * DIMC + c], yw);
}

// ------------------------------------- per-axis feature pipeline (48 rows)
// pooled @ Win @ prW -> LN -> GeAct MLP -> latent (3,48,64)
__global__ void k_axis_feat(const float* poolx, const float* pooly, const float* poolz,
                            const float* Win, const float* prW,
                            const float* pr_lnw, const float* pr_lnb,
                            const float* mlpW1, const float* mlpb1,
                            const float* mlpW2, const float* mlpb2,
                            float* ux) {
  int axis = blockIdx.x / 48, row = blockIdx.x % 48;
  const float* pool = axis == 0 ? poolx : (axis == 1 ? pooly : poolz);
  __shared__ float sA[128], sB[128], red[128];
  int t = threadIdx.x;
  sA[t] = pool[row * 128 + t];
  __syncthreads();
  float acc = 0.f;
  for (int k = 0; k < 128; ++k) acc += sA[k] * Win[k * 128 + t];
  __syncthreads();
  sB[t] = acc; __syncthreads();
  const float* Wp = prW + axis * 128 * 128;
  acc = 0.f;
  for (int k = 0; k < 128; ++k) acc += sB[k] * Wp[k * 128 + t];
  __syncthreads();
  sA[t] = acc; __syncthreads();
  // LayerNorm over 128
  red[t] = sA[t]; __syncthreads();
  for (int s = 64; s > 0; s >>= 1) { if (t < s) red[t] += red[t + s]; __syncthreads(); }
  float mean = red[0] * (1.0f / 128.f); __syncthreads();
  float dv = sA[t] - mean;
  red[t] = dv * dv; __syncthreads();
  for (int s = 64; s > 0; s >>= 1) { if (t < s) red[t] += red[t + s]; __syncthreads(); }
  float var = red[0] * (1.0f / 128.f);
  float h = dv * rsqrtf(var + 1e-5f) * pr_lnw[axis * 128 + t] + pr_lnb[axis * 128 + t];
  __syncthreads();
  sB[t] = h; __syncthreads();
  // Linear(128 -> 256) + GeAct: gelu(a)*g
  const float* W1 = mlpW1 + axis * 128 * 256;
  const float* b1 = mlpb1 + axis * 256;
  float a = b1[t], g = b1[128 + t];
  for (int k = 0; k < 128; ++k) {
    float s = sB[k];
    a += s * W1[k * 256 + t];
    g += s * W1[k * 256 + 128 + t];
  }
  float act = gelu_exact(a) * g;
  __syncthreads();
  sA[t] = act; __syncthreads();
  if (t < 64) {
    const float* W2 = mlpW2 + axis * 128 * 64;
    float o = mlpb2[axis * 64 + t];
    for (int k = 0; k < 128; ++k) o += sA[k] * W2[k * 64 + t];
    ux[(axis * 48 + row) * 64 + t] = o;
  }
}

// ----------------------------------------- q/k projection + RoPE (per row)
__global__ void k_qk_rope(const float* __restrict__ ux,
                          const float* __restrict__ Wq, const float* __restrict__ Wk,
                          const float* pos_x, const float* pos_y, const float* pos_z,
                          float* __restrict__ qbuf, float* __restrict__ kbuf) {
  int axis = blockIdx.x / 48, row = blockIdx.x % 48;
  const float* pos = axis == 0 ? pos_x : (axis == 1 ? pos_y : pos_z);
  __shared__ float uxr[64], qs[1536], ks[1536];
  int t = threadIdx.x;  // 256
  if (t < 64) uxr[t] = ux[(axis * 48 + row) * 64 + t];
  __syncthreads();
  const float* WqA = Wq + (size_t)axis * 64 * 1536;
  const float* WkA = Wk + (size_t)axis * 64 * 1536;
  for (int o = t; o < 1536; o += 256) {
    float aq = 0.f, ak = 0.f;
    for (int l = 0; l < 64; ++l) {
      float s = uxr[l];
      aq += s * WqA[l * 1536 + o];
      ak += s * WkA[l * 1536 + o];
    }
    qs[o] = aq; ks[o] = ak;
  }
  __syncthreads();
  float tpos = pos[row] * 64.0f;  // pos / MIN_FREQ
  for (int o = t; o < 1536; o += 256) {
    int h = o / 192, d = o % 192;
    int fi = d % 96;
    float f = tpos * powf(10000.0f, -((float)(2 * fi)) / 192.0f);
    float cf = cosf(f), sf = sinf(f);
    int base = h * 192;
    float qrot = (d < 96) ? -qs[base + d + 96] : qs[base + d - 96];
    float krot = (d < 96) ? -ks[base + d + 96] : ks[base + d - 96];
    size_t idx = ((size_t)(axis * 48 + row)) * 1536 + o;
    qbuf[idx] = qs[o] * cf + qrot * sf;
    kbuf[idx] = ks[o] * cf + krot * sf;
  }
}

// -------------------------- K-matrices: K[h,i,j] = q_i . k_j  (bf16, j padded to 64)
__global__ void k_kmat(const float* __restrict__ qbuf, const float* __restrict__ kbuf,
                       __bf16* __restrict__ kmat) {
  int axis = blockIdx.x / 8, h = blockIdx.x % 8;
  __shared__ float ks[48 * 192];
  int t = threadIdx.x;  // 256
  for (int idx = t; idx < 48 * 192; idx += 256) {
    int j = idx / 192, d = idx % 192;
    ks[idx] = kbuf[((size_t)(axis * 48 + j)) * 1536 + h * 192 + d];
  }
  __syncthreads();
  for (int oi = t; oi < 48 * 48; oi += 256) {
    int i = oi / 48, j = oi % 48;
    const float* qp = qbuf + ((size_t)(axis * 48 + i)) * 1536 + h * 192;
    const float* kp = ks + j * 192;
    float acc = 0.f;
    for (int d = 0; d < 192; ++d) acc += qp[d] * kp[d];
    kmat[(((size_t)axis * 8 + h) * 48 + i) * 64 + j] = f2bf(acc);
  }
  for (int idx = t; idx < 48 * 16; idx += 256) {   // zero-pad j = 48..63
    int i = idx / 16, j = 48 + (idx % 16);
    kmat[(((size_t)axis * 8 + h) * 48 + i) * 64 + j] = f2bf(0.f);
  }
}

// ---------------------------------------------------- generic WMMA bf16 GEMM
// A: (M,K) bf16 row-major; Bt: (N,K) bf16 row-major (= B transposed);
// one 16x16 C tile per wave; blockDim = (32,4).
__global__ void k_wmma_gemm(const __bf16* __restrict__ A, const __bf16* __restrict__ Bt,
                            void* __restrict__ C, int M, int N, int K,
                            int outBf16, int doGelu) {
  int ntiles = N >> 4;
  int tile = blockIdx.x * blockDim.y + threadIdx.y;
  int mt = tile / ntiles, nt = tile - mt * ntiles;
  if (mt * 16 >= M) return;
  int lane = threadIdx.x, l16 = lane & 15, g = lane >> 4;
  const __bf16* Ar = A + (size_t)(mt * 16 + l16) * K;   // row m
  const __bf16* Br = Bt + (size_t)(nt * 16 + l16) * K;  // column n (transposed)
  v8f acc = {};
  for (int k0 = 0; k0 < K; k0 += 32) {
    v16bf af, bfv;
    const __bf16* a0 = Ar + k0 + 16 * g;  // A frag: K = k0+16g+e
#pragma unroll
    for (int e = 0; e < 8; ++e) { af[e] = a0[e]; af[e + 8] = a0[8 + e]; }
    const __bf16* b0 = Br + k0 + 16 * g;  // B frag: K = k0+16g+e
#pragma unroll
    for (int e = 0; e < 16; ++e) bfv[e] = b0[e];
    acc = __builtin_amdgcn_wmma_f32_16x16x32_bf16(false, af, false, bfv,
                                                  (short)0, acc, false, false);
  }
  int n = nt * 16 + l16;
#pragma unroll
  for (int r = 0; r < 8; ++r) {           // D: M = r + 8g, N = l16
    int m = mt * 16 + r + 8 * g;
    float v = acc[r];
    if (doGelu) v = gelu_exact(v);
    if (outBf16) ((__bf16*)C)[(size_t)m * N + n] = f2bf(v);
    else         ((float*)C)[(size_t)m * N + n] = v;
  }
}

// ------------------------- factorized contraction along one spatial axis
// O[.., i, ..][h,c] = sum_j Km[h,i,j] * T[.., j, ..][h,c]
// T/O layout (x,y,z,h,c) bf16; strideA = contracted axis stride, st1/st2 = others.
__global__ void k_contract(const __bf16* __restrict__ Km, const __bf16* __restrict__ T,
                           __bf16* __restrict__ O, long strideA, long st1, long st2) {
  int h = blockIdx.z;
  int it = blockIdx.y;
  int ntile = blockIdx.x * blockDim.y + threadIdx.y;
  int lane = threadIdx.x, l16 = lane & 15, g = lane >> 4;
  int n0 = ntile << 4;          // n = (s1*48 + s2)*64 + c ; 16-wide tile stays in one c-block
  int yz = n0 >> 6;
  int cbase = n0 & 63;
  int s1 = yz / 48, s2 = yz % 48;
  long colOff = (long)s1 * st1 + (long)s2 * st2 + (long)h * 64 + cbase + l16;
  const __bf16* Kh = Km + ((size_t)h * 48 + it * 16 + l16) * 64;  // A row i, jpad=64
  v8f acc = {};
#pragma unroll
  for (int k0 = 0; k0 < 64; k0 += 32) {
    v16bf af, bfv;
    const __bf16* a0 = Kh + k0 + 16 * g;
#pragma unroll
    for (int e = 0; e < 8; ++e) { af[e] = a0[e]; af[e + 8] = a0[8 + e]; }
#pragma unroll
    for (int e = 0; e < 16; ++e) {
      int j = k0 + 16 * g + e;
      bfv[e] = (j < 48) ? T[colOff + (long)j * strideA] : f2bf(0.f);
    }
    acc = __builtin_amdgcn_wmma_f32_16x16x32_bf16(false, af, false, bfv,
                                                  (short)0, acc, false, false);
  }
#pragma unroll
  for (int r = 0; r < 8; ++r) {
    int i = it * 16 + r + 8 * g;
    O[colOff + (long)i * strideA] = f2bf(acc[r]);
  }
}

// --------------------------------------------- instance norm (per-x group)
__global__ void k_inorm_stats(const __bf16* __restrict__ T, float* stats) {
  __shared__ float rs[256], rq[256];
  int x = blockIdx.x, t = threadIdx.x;
  const __bf16* p = T + (size_t)x * (48 * 48 * 512);
  float s = 0.f, q = 0.f;
  for (int i = t; i < 48 * 48 * 512; i += 256) {
    float v = bf2f(p[i]); s += v; q += v * v;
  }
  rs[t] = s; rq[t] = q; __syncthreads();
  for (int st = 128; st > 0; st >>= 1) {
    if (t < st) { rs[t] += rs[t + st]; rq[t] += rq[t + st]; }
    __syncthreads();
  }
  if (t == 0) {
    float n = 48.f * 48.f * 512.f;
    float mean = rs[0] / n;
    float var = rq[0] / n - mean * mean;
    stats[x * 2] = mean;
    stats[x * 2 + 1] = rsqrtf(var + 1e-5f);
  }
}

__global__ void k_inorm_apply(__bf16* __restrict__ T, const float* __restrict__ stats) {
  size_t total = (size_t)NPTS * 512;
  for (size_t i = (size_t)blockIdx.x * blockDim.x + threadIdx.x; i < total;
       i += (size_t)gridDim.x * blockDim.x) {
    int x = (int)(i / (48 * 48 * 512));
    float v = (bf2f(T[i]) - stats[x * 2]) * stats[x * 2 + 1];
    T[i] = f2bf(v);
  }
}

// --------------------------------------------------------------------------
extern "C" void kernel_launch(void* const* d_in, const int* in_sizes, int n_in,
                              void* d_out, int out_size, void* d_ws, size_t ws_size,
                              hipStream_t stream) {
  (void)in_sizes; (void)n_in; (void)out_size; (void)ws_size;
  const float* u     = (const float*)d_in[0];
  const float* pos_x = (const float*)d_in[1];
  const float* pos_y = (const float*)d_in[2];
  const float* pos_z = (const float*)d_in[3];
  const float* lnw   = (const float*)d_in[4];
  const float* lnb   = (const float*)d_in[5];
  const float* Wv    = (const float*)d_in[6];
  const float* Win   = (const float*)d_in[7];
  const float* prW   = (const float*)d_in[8];
  const float* prlnw = (const float*)d_in[9];
  const float* prlnb = (const float*)d_in[10];
  const float* mlpW1 = (const float*)d_in[11];
  const float* mlpb1 = (const float*)d_in[12];
  const float* mlpW2 = (const float*)d_in[13];
  const float* mlpb2 = (const float*)d_in[14];
  const float* Wq    = (const float*)d_in[15];
  const float* Wk    = (const float*)d_in[16];
  const float* outW1 = (const float*)d_in[17];
  const float* outW2 = (const float*)d_in[18];

  size_t off = 0;
  auto carve = [&](size_t bytes) {
    char* p = (char*)d_ws + off;
    off += (bytes + 255) & ~(size_t)255;
    return p;
  };
  __bf16* lnU   = (__bf16*)carve((size_t)NPTS * 128 * 2);   // ln(u) bf16; reused as MLP mid
  __bf16* bufA  = (__bf16*)carve((size_t)NPTS * 512 * 2);   // v / t2
  __bf16* bufB  = (__bf16*)carve((size_t)NPTS * 512 * 2);   // t1 / t3
  __bf16* WvT   = (__bf16*)carve((size_t)512 * 128 * 2);
  __bf16* W1T   = (__bf16*)carve((size_t)128 * 512 * 2);
  __bf16* W2T   = (__bf16*)carve((size_t)128 * 128 * 2);
  float*  poolx = (float*)carve(48 * 128 * 4);
  float*  pooly = (float*)carve(48 * 128 * 4);
  float*  poolz = (float*)carve(48 * 128 * 4);
  float*  ux    = (float*)carve(3 * 48 * 64 * 4);
  float*  qbuf  = (float*)carve((size_t)3 * 48 * 1536 * 4);
  float*  kbuf  = (float*)carve((size_t)3 * 48 * 1536 * 4);
  __bf16* kmat  = (__bf16*)carve((size_t)3 * 8 * 48 * 64 * 2);
  float*  stats = (float*)carve(48 * 2 * 4);

  // zero pooled accumulators
  k_zero<<<dim3(24), dim3(256), 0, stream>>>(poolx, 48 * 128);
  k_zero<<<dim3(24), dim3(256), 0, stream>>>(pooly, 48 * 128);
  k_zero<<<dim3(24), dim3(256), 0, stream>>>(poolz, 48 * 128);

  // weight transposes -> bf16
  k_convT<<<dim3((128 * 512 + 255) / 256), dim3(256), 0, stream>>>(Wv, WvT, 128, 512);
  k_convT<<<dim3((512 * 128 + 255) / 256), dim3(256), 0, stream>>>(outW1, W1T, 512, 128);
  k_convT<<<dim3((128 * 128 + 255) / 256), dim3(256), 0, stream>>>(outW2, W2T, 128, 128);

  // LayerNorm + pooled means
  k_ln_pool<<<dim3(NPTS), dim3(128), 0, stream>>>(u, lnw, lnb, lnU, poolx, pooly, poolz);

  // per-axis latent features, q/k + rope, K-matrices
  k_axis_feat<<<dim3(3 * 48), dim3(128), 0, stream>>>(poolx, pooly, poolz, Win, prW,
                                                      prlnw, prlnb, mlpW1, mlpb1,
                                                      mlpW2, mlpb2, ux);
  k_qk_rope<<<dim3(3 * 48), dim3(256), 0, stream>>>(ux, Wq, Wk, pos_x, pos_y, pos_z,
                                                    qbuf, kbuf);
  k_kmat<<<dim3(24), dim3(256), 0, stream>>>(qbuf, kbuf, kmat);

  // v = ln(u) @ Wv   (WMMA)
  {
    int M = NPTS, N = 512, K = 128;
    int tiles = (M / 16) * (N / 16);
    k_wmma_gemm<<<dim3(tiles / 4), dim3(32, 4), 0, stream>>>(lnU, WvT, bufA, M, N, K, 1, 0);
  }

  // factorized contractions along x, y, z (WMMA)
  long sx = 48L * 48 * 512, sy = 48L * 512, sz = 512;
  dim3 cgrid(9216 / 4, 3, 8), cblk(32, 4);
  k_contract<<<cgrid, cblk, 0, stream>>>(kmat + (size_t)0 * 8 * 48 * 64, bufA, bufB, sx, sy, sz);
  k_contract<<<cgrid, cblk, 0, stream>>>(kmat + (size_t)1 * 8 * 48 * 64, bufB, bufA, sy, sx, sz);
  k_contract<<<cgrid, cblk, 0, stream>>>(kmat + (size_t)2 * 8 * 48 * 64, bufA, bufB, sz, sx, sy);

  // instance norm over (ny,nz,hc) per (b,nx)
  k_inorm_stats<<<dim3(48), dim3(256), 0, stream>>>(bufB, stats);
  k_inorm_apply<<<dim3(4096), dim3(256), 0, stream>>>(bufB, stats);

  // out = gelu(h @ outW1) @ outW2   (two WMMA GEMMs)
  {
    int M = NPTS, N = 128, K = 512;
    int tiles = (M / 16) * (N / 16);
    k_wmma_gemm<<<dim3(tiles / 4), dim3(32, 4), 0, stream>>>(bufB, W1T, lnU, M, N, K, 1, 1);
  }
  {
    int M = NPTS, N = 128, K = 128;
    int tiles = (M / 16) * (N / 16);
    k_wmma_gemm<<<dim3(tiles / 4), dim3(32, 4), 0, stream>>>(lnU, W2T, d_out, M, N, K, 0, 0);
  }
}
